// LSTM_16432544875406
// MI455X (gfx1250) — matrix operand order repaired and checked by reading, not compiled, not run on hardware
//
#include <hip/hip_runtime.h>
#include <math.h>

#define SEQ   512
#define D_IN  256
#define D_H   512
#define NCLS  1024
#define BB    256
#define KTOT  768     // D_IN + D_H
#define NG    2048    // 4 * D_H
#define KT_X    8     // D_IN / 32
#define KT_H   16     // D_H  / 32
#define KT_STEP 24    // KTOT / 32
#define KT_PROJ 16    // D_H  / 32

typedef __attribute__((ext_vector_type(16))) __bf16 bf16x16;
typedef __attribute__((ext_vector_type(8)))  __bf16 bf16x8;
typedef __attribute__((ext_vector_type(8)))  float  f32x8;

// ---------------------------------------------------------------- helpers

__device__ __forceinline__ float sigmoidf_(float v) {
    return 1.0f / (1.0f + __expf(-v));
}

// Load one 16-bit A fragment (16x32, row-major source) for this lane.
// Layout (ISA 7.12.2): lanes 0-15 hold row M=lane, K=[0..7] in v0-3 and
// K=[16..23] in v4-7; lanes 16-31 hold row M=lane-16, K=[8..15],[24..31].
__device__ __forceinline__ bf16x16 load_a_frag(const __bf16* row_ptr /* at (row, kbase+koff) */) {
    bf16x8 lo = *(const bf16x8*)row_ptr;
    bf16x8 hi = *(const bf16x8*)(row_ptr + 16);
    bf16x16 a;
#pragma unroll
    for (int i = 0; i < 8; ++i) { a[i] = lo[i]; a[i + 8] = hi[i]; }
    return a;
}

__device__ __forceinline__ void mma4(const bf16x16& a0, const bf16x16& a1,
                                     const bf16x16& b0, const bf16x16& b1,
                                     f32x8& c00, f32x8& c01, f32x8& c10, f32x8& c11) {
    c00 = __builtin_amdgcn_wmma_f32_16x16x32_bf16(false, a0, false, b0, (short)0, c00, false, false);
    c01 = __builtin_amdgcn_wmma_f32_16x16x32_bf16(false, a0, false, b1, (short)0, c01, false, false);
    c10 = __builtin_amdgcn_wmma_f32_16x16x32_bf16(false, a1, false, b0, (short)0, c10, false, false);
    c11 = __builtin_amdgcn_wmma_f32_16x16x32_bf16(false, a1, false, b1, (short)0, c11, false, false);
}

// ---------------------------------------------------------------- prep kernels

__global__ __launch_bounds__(256) void convert_x_bf16(const float* __restrict__ x,
                                                      __bf16* __restrict__ xb, int n) {
    int idx = blockIdx.x * 256 + threadIdx.x;
    if (idx < n) xb[idx] = (__bf16)x[idx];
}

// Pack combined [Wx;Wh] (K=768, N=2048) into WMMA B-fragment order:
// wpack[((nt*KT_STEP + kt)*32 + lane)*16 + e] = W[kt*32 + (lane<16?0:8) + (e<8?e:8+e)][nt*16 + lane%16]
__global__ __launch_bounds__(256) void pack_w_gates(
        const float* __restrict__ Wgx, const float* __restrict__ Wgh,
        const float* __restrict__ Wix, const float* __restrict__ Wih,
        const float* __restrict__ Wfx, const float* __restrict__ Wfh,
        const float* __restrict__ Wox, const float* __restrict__ Woh,
        __bf16* __restrict__ wpack) {
    int tid = blockIdx.x * 256 + threadIdx.x;            // KTOT*NG threads
    int e    = tid & 15;
    int lane = (tid >> 4) & 31;
    int rest = tid >> 9;
    int kt   = rest % KT_STEP;
    int nt   = rest / KT_STEP;
    int n = (nt << 4) + (lane & 15);
    int k = (kt << 5) + ((lane < 16) ? 0 : 8) + ((e < 8) ? e : 8 + e);
    int gate = n >> 9;           // 0..3 -> g,i,f,o
    int col  = n & (D_H - 1);
    const float* Wx[4] = {Wgx, Wix, Wfx, Wox};
    const float* Wh[4] = {Wgh, Wih, Wfh, Woh};
    float v = (k < D_IN) ? Wx[gate][(size_t)k * D_H + col]
                         : Wh[gate][(size_t)(k - D_IN) * D_H + col];
    wpack[tid] = (__bf16)v;
}

__global__ __launch_bounds__(256) void pack_w_proj(const float* __restrict__ Wph,
                                                   __bf16* __restrict__ wppack) {
    int tid = blockIdx.x * 256 + threadIdx.x;            // D_H*NCLS threads
    int e    = tid & 15;
    int lane = (tid >> 4) & 31;
    int rest = tid >> 9;
    int kt   = rest % KT_PROJ;
    int nt   = rest / KT_PROJ;
    int n = (nt << 4) + (lane & 15);
    int k = (kt << 5) + ((lane < 16) ? 0 : 8) + ((e < 8) ? e : 8 + e);
    wppack[tid] = (__bf16)Wph[(size_t)k * NCLS + n];
}

__global__ __launch_bounds__(256) void combine_bias(
        const float* __restrict__ bg, const float* __restrict__ bi,
        const float* __restrict__ bf, const float* __restrict__ bo,
        float* __restrict__ bias) {
    int n = blockIdx.x * 256 + threadIdx.x;              // NG threads
    const float* b4[4] = {bg, bi, bf, bo};
    bias[n] = b4[n >> 9][n & (D_H - 1)];
}

__global__ __launch_bounds__(256) void zero_state(float* __restrict__ c,
                                                  __bf16* __restrict__ hb) {
    int idx = blockIdx.x * 256 + threadIdx.x;            // BB*D_H threads
    c[idx]  = 0.0f;
    hb[idx] = (__bf16)0.0f;
}

// ---------------------------------------------------------------- per-step GEMM
// gates = act([x_t | h_{t-1}] @ [Wx;Wh] + bias).
// Grid: (NG/128, BB/64), 256 threads = 8 waves; wave -> 32x32 output tile.
// K loop split into x segment (8 tiles) and h segment (16 tiles): no per-iter
// branch, and all per-lane base addresses hoisted so the inner loops are pure
// immediate-offset global_load_b128 + v_wmma.
__global__ __launch_bounds__(256) void lstm_step_gemm(
        const __bf16* __restrict__ xb, const __bf16* __restrict__ hb,
        const __bf16* __restrict__ wpack, const float* __restrict__ bias,
        float* __restrict__ gates, int t) {
    const int lane = threadIdx.x & 31;
    const int wave = threadIdx.x >> 5;
    const int wm = wave >> 2, wn = wave & 3;
    const int m0 = blockIdx.y * 64 + wm * 32;
    const int n0 = blockIdx.x * 128 + wn * 32;
    const int nt0 = n0 >> 4;

    const int lr   = lane & 15;
    const int koff = (lane < 16) ? 0 : 8;

    f32x8 c00 = {}, c01 = {}, c10 = {}, c11 = {};

    // Hoisted per-lane base pointers.
    const __bf16* pax0 = xb + ((size_t)(m0 + lr) * SEQ + t) * D_IN + koff;
    const __bf16* pax1 = pax0 + (size_t)16 * SEQ * D_IN;
    const __bf16* pah0 = hb + (size_t)(m0 + lr) * D_H + koff;
    const __bf16* pah1 = pah0 + (size_t)16 * D_H;
    const __bf16* pb0  = wpack + (((size_t)nt0 * KT_STEP) * 32 + lane) * 16;
    const __bf16* pb1  = pb0 + (size_t)KT_STEP * 512;      // next n-tile

    // x segment: K = 0..255
#pragma unroll
    for (int kt = 0; kt < KT_X; ++kt) {
        bf16x16 a0 = load_a_frag(pax0 + kt * 32);
        bf16x16 a1 = load_a_frag(pax1 + kt * 32);
        bf16x16 b0 = *(const bf16x16*)(pb0 + kt * 512);
        bf16x16 b1 = *(const bf16x16*)(pb1 + kt * 512);
        mma4(a0, a1, b0, b1, c00, c01, c10, c11);
    }
    // h segment: K = 256..767
#pragma unroll
    for (int kt = 0; kt < KT_H; ++kt) {
        bf16x16 a0 = load_a_frag(pah0 + kt * 32);
        bf16x16 a1 = load_a_frag(pah1 + kt * 32);
        bf16x16 b0 = *(const bf16x16*)(pb0 + (KT_X + kt) * 512);
        bf16x16 b1 = *(const bf16x16*)(pb1 + (KT_X + kt) * 512);
        mma4(a0, a1, b0, b1, c00, c01, c10, c11);
    }

    // Epilogue: bias + (sigmoid for gate 'g', tanh for i/f/o — reference inverts them).
    const int colA = n0 + lr;
    const int colB = n0 + 16 + lr;
    const int mo = (lane < 16) ? 0 : 8;
    const float bA = bias[colA], bB = bias[colB];
    const bool sigA = (colA >> 9) == 0;
    const bool sigB = (colB >> 9) == 0;
#pragma unroll
    for (int r = 0; r < 8; ++r) {
        int mA = m0 + mo + r;
        int mB = m0 + 16 + mo + r;
        float v;
        v = c00[r] + bA; v = sigA ? sigmoidf_(v) : tanhf(v); gates[(size_t)mA * NG + colA] = v;
        v = c01[r] + bB; v = sigB ? sigmoidf_(v) : tanhf(v); gates[(size_t)mA * NG + colB] = v;
        v = c10[r] + bA; v = sigA ? sigmoidf_(v) : tanhf(v); gates[(size_t)mB * NG + colA] = v;
        v = c11[r] + bB; v = sigB ? sigmoidf_(v) : tanhf(v); gates[(size_t)mB * NG + colB] = v;
    }
}

__global__ __launch_bounds__(256) void lstm_cell_update(const float* __restrict__ gates,
                                                        float* __restrict__ c,
                                                        __bf16* __restrict__ hb) {
    int idx = blockIdx.x * 256 + threadIdx.x;            // BB*D_H threads
    int b = idx >> 9, j = idx & (D_H - 1);
    const float* gr = gates + (size_t)b * NG;
    float g = gr[j];
    float i = gr[D_H + j];
    float f = gr[2 * D_H + j];
    float o = gr[3 * D_H + j];
    float cn = g * i + c[idx] * f;
    c[idx] = cn;
    hb[idx] = (__bf16)(tanhf(cn) * o);
}

// ---------------------------------------------------------------- projection + softmax

__global__ __launch_bounds__(256) void proj_gemm(
        const __bf16* __restrict__ hb, const __bf16* __restrict__ wppack,
        const float* __restrict__ bp, float* __restrict__ logits) {
    const int lane = threadIdx.x & 31;
    const int wave = threadIdx.x >> 5;
    const int wm = wave >> 2, wn = wave & 3;
    const int m0 = blockIdx.y * 64 + wm * 32;
    const int n0 = blockIdx.x * 128 + wn * 32;
    const int nt0 = n0 >> 4;

    const int lr   = lane & 15;
    const int koff = (lane < 16) ? 0 : 8;

    f32x8 c00 = {}, c01 = {}, c10 = {}, c11 = {};

    const __bf16* pa0 = hb + (size_t)(m0 + lr) * D_H + koff;
    const __bf16* pa1 = pa0 + (size_t)16 * D_H;
    const __bf16* pb0 = wppack + (((size_t)nt0 * KT_PROJ) * 32 + lane) * 16;
    const __bf16* pb1 = pb0 + (size_t)KT_PROJ * 512;

#pragma unroll
    for (int kt = 0; kt < KT_PROJ; ++kt) {
        bf16x16 a0 = load_a_frag(pa0 + kt * 32);
        bf16x16 a1 = load_a_frag(pa1 + kt * 32);
        bf16x16 b0 = *(const bf16x16*)(pb0 + kt * 512);
        bf16x16 b1 = *(const bf16x16*)(pb1 + kt * 512);
        mma4(a0, a1, b0, b1, c00, c01, c10, c11);
    }

    const int colA = n0 + lr;
    const int colB = n0 + 16 + lr;
    const int mo = (lane < 16) ? 0 : 8;
    const float bA = bp[colA], bB = bp[colB];
#pragma unroll
    for (int r = 0; r < 8; ++r) {
        int mA = m0 + mo + r;
        int mB = m0 + 16 + mo + r;
        logits[(size_t)mA * NCLS + colA] = c00[r] + bA;
        logits[(size_t)mA * NCLS + colB] = c01[r] + bB;
        logits[(size_t)mB * NCLS + colA] = c10[r] + bA;
        logits[(size_t)mB * NCLS + colB] = c11[r] + bB;
    }
}

__global__ __launch_bounds__(256) void softmax_rows(const float* __restrict__ logits,
                                                    float* __restrict__ out) {
    __shared__ float red[256];
    const int row = blockIdx.x;
    const float* lp = logits + (size_t)row * NCLS;

    float lmax = -INFINITY;
    for (int c = threadIdx.x; c < NCLS; c += 256) lmax = fmaxf(lmax, lp[c]);
    red[threadIdx.x] = lmax; __syncthreads();
    for (int s = 128; s > 0; s >>= 1) {
        if (threadIdx.x < s) red[threadIdx.x] = fmaxf(red[threadIdx.x], red[threadIdx.x + s]);
        __syncthreads();
    }
    lmax = red[0]; __syncthreads();

    float lsum = 0.0f;
    for (int c = threadIdx.x; c < NCLS; c += 256) lsum += __expf(lp[c] - lmax);
    red[threadIdx.x] = lsum; __syncthreads();
    for (int s = 128; s > 0; s >>= 1) {
        if (threadIdx.x < s) red[threadIdx.x] += red[threadIdx.x + s];
        __syncthreads();
    }
    float inv = 1.0f / red[0];

    for (int c = threadIdx.x; c < NCLS; c += 256)
        out[(size_t)row * NCLS + c] = __expf(lp[c] - lmax) * inv;
}

// ---------------------------------------------------------------- launcher

extern "C" void kernel_launch(void* const* d_in, const int* in_sizes, int n_in,
                              void* d_out, int out_size, void* d_ws, size_t ws_size,
                              hipStream_t stream) {
    const float* x   = (const float*)d_in[0];
    const float* Wgx = (const float*)d_in[1];
    const float* Wgh = (const float*)d_in[2];
    const float* bg  = (const float*)d_in[3];
    const float* Wix = (const float*)d_in[4];
    const float* Wih = (const float*)d_in[5];
    const float* bi  = (const float*)d_in[6];
    const float* Wfx = (const float*)d_in[7];
    const float* Wfh = (const float*)d_in[8];
    const float* bf  = (const float*)d_in[9];
    const float* Wox = (const float*)d_in[10];
    const float* Woh = (const float*)d_in[11];
    const float* bo  = (const float*)d_in[12];
    const float* Wph = (const float*)d_in[13];
    const float* bp  = (const float*)d_in[14];

    char* ws = (char*)d_ws;
    size_t off = 0;
    auto carve = [&](size_t bytes) -> char* {
        char* p = ws + off;
        off = (off + bytes + 255) & ~(size_t)255;
        return p;
    };
    __bf16* xb     = (__bf16*)carve((size_t)BB * SEQ * D_IN * 2);   // 64 MB
    __bf16* wpack  = (__bf16*)carve((size_t)KTOT * NG * 2);         // 3 MB
    __bf16* wppack = (__bf16*)carve((size_t)D_H * NCLS * 2);        // 1 MB
    float*  bias   = (float*)carve((size_t)NG * 4);
    float*  cbuf   = (float*)carve((size_t)BB * D_H * 4);
    __bf16* hb     = (__bf16*)carve((size_t)BB * D_H * 2);
    float*  gates  = (float*)carve((size_t)BB * NG * 4);            // 2 MB
    float*  logits = (float*)carve((size_t)BB * NCLS * 4);          // 1 MB

    // One-time prep (per launch; deterministic).
    convert_x_bf16<<<(BB * SEQ * D_IN) / 256, 256, 0, stream>>>(x, xb, BB * SEQ * D_IN);
    pack_w_gates<<<(KTOT * NG) / 256, 256, 0, stream>>>(Wgx, Wgh, Wix, Wih, Wfx, Wfh, Wox, Woh, wpack);
    pack_w_proj<<<(D_H * NCLS) / 256, 256, 0, stream>>>(Wph, wppack);
    combine_bias<<<NG / 256, 256, 0, stream>>>(bg, bi, bf, bo, bias);
    zero_state<<<(BB * D_H) / 256, 256, 0, stream>>>(cbuf, hb);

    // Recurrence.
    dim3 gStep(NG / 128, BB / 64);      // (16, 4)
    for (int t = 0; t < SEQ; ++t) {
        lstm_step_gemm<<<gStep, 256, 0, stream>>>(xb, hb, wpack, bias, gates, t);
        lstm_cell_update<<<(BB * D_H) / 256, 256, 0, stream>>>(gates, cbuf, hb);
    }

    // Projection + softmax.
    dim3 gProj(NCLS / 128, BB / 64);    // (8, 4)
    proj_gemm<<<gProj, 256, 0, stream>>>(hb, wppack, bp, logits);
    softmax_rows<<<BB, 256, 0, stream>>>(logits, (float*)d_out);

    (void)in_sizes; (void)n_in; (void)out_size; (void)ws_size;
}
